// Alert_82463372083842
// MI455X (gfx1250) — compile-verified, matrix operand order checked by reading, not run on hardware
//
#include <hip/hip_runtime.h>
#include <math.h>

typedef __attribute__((ext_vector_type(16))) _Float16 v16h;
typedef __attribute__((ext_vector_type(8)))  float    v8f;
typedef __attribute__((ext_vector_type(4)))  float    f4;

#define HW   (1024 * 1024)
#define NCH  384
#define NCH4 96          // NCH / 4 (float4 columns per row)

// ---------------------------------------------------------------------------
// Kernel 1: streaming per-channel partial reduction (sum, sumsq, max).
// Block = 384 threads. Thread t: q = t%96 (float4 column), rs = t/96 (row
// subgroup). Each thread streams rows {base+rs, base+rs+4, ...} with b128
// non-temporal loads, then the 4 row subgroups are combined through LDS.
// ---------------------------------------------------------------------------
__global__ void pool_partial_kernel(const f4* __restrict__ x4,
                                    float* __restrict__ psum,
                                    float* __restrict__ psq,
                                    float* __restrict__ pmax,
                                    int rows_per_block) {
  __shared__ float red[3][4][NCH];   // [stat][row-subgroup][channel]

  const int t  = threadIdx.x;
  const int q  = t % NCH4;
  const int rs = t / NCH4;
  const long rowBase = (long)blockIdx.x * rows_per_block + rs;

  f4 s  = {0.f, 0.f, 0.f, 0.f};
  f4 ss = {0.f, 0.f, 0.f, 0.f};
  f4 mx = {-3.402823466e38f, -3.402823466e38f, -3.402823466e38f, -3.402823466e38f};

  const int iters = rows_per_block >> 2;
  for (int i = 0; i < iters; ++i) {
    const long row = rowBase + 4L * i;
    f4 v = __builtin_nontemporal_load(&x4[row * NCH4 + q]);
    s  += v;
    ss += v * v;
    mx.x = fmaxf(mx.x, v.x);
    mx.y = fmaxf(mx.y, v.y);
    mx.z = fmaxf(mx.z, v.z);
    mx.w = fmaxf(mx.w, v.w);
  }

  const int c0 = q * 4;
  red[0][rs][c0 + 0] = s.x;  red[0][rs][c0 + 1] = s.y;
  red[0][rs][c0 + 2] = s.z;  red[0][rs][c0 + 3] = s.w;
  red[1][rs][c0 + 0] = ss.x; red[1][rs][c0 + 1] = ss.y;
  red[1][rs][c0 + 2] = ss.z; red[1][rs][c0 + 3] = ss.w;
  red[2][rs][c0 + 0] = mx.x; red[2][rs][c0 + 1] = mx.y;
  red[2][rs][c0 + 2] = mx.z; red[2][rs][c0 + 3] = mx.w;
  __syncthreads();

  // thread t now owns channel t
  const float ts = red[0][0][t] + red[0][1][t] + red[0][2][t] + red[0][3][t];
  const float tq = red[1][0][t] + red[1][1][t] + red[1][2][t] + red[1][3][t];
  const float tm = fmaxf(fmaxf(red[2][0][t], red[2][1][t]),
                         fmaxf(red[2][2][t], red[2][3][t]));
  const long o = (long)blockIdx.x * NCH + t;
  psum[o] = ts;
  psq[o]  = tq;
  pmax[o] = tm;
}

// ---------------------------------------------------------------------------
// Kernel 2: fold partials -> feats[1152] in LDS, then 3-layer MLP.
// Layer 1 (1152 -> 64) runs on WMMA: waves 0..3 each own a 16-column slice of
// W1; A holds feats replicated across all 16 rows (layout is M-independent),
// so D row 0 (acc[0], lanes 0..15) is the GEMV result.
// ---------------------------------------------------------------------------
__global__ void finalize_mlp_kernel(const float* __restrict__ psum,
                                    const float* __restrict__ psq,
                                    const float* __restrict__ pmax,
                                    int nblocks,
                                    const float* __restrict__ W1,
                                    const float* __restrict__ b1,
                                    const float* __restrict__ W2,
                                    const float* __restrict__ b2,
                                    const float* __restrict__ W3,
                                    const float* __restrict__ b3,
                                    float* __restrict__ out) {
  __shared__ float feats[3 * NCH];
  __shared__ float h1[64];
  __shared__ float h2v[16];

  const int t = threadIdx.x;

  // ---- Phase 1: combine partials; feats = [mean | max | std] ----
  {
    float s = 0.f, sq = 0.f, m = -3.402823466e38f;
    for (int b = 0; b < nblocks; ++b) {
      const long o = (long)b * NCH + t;
      s  += psum[o];
      sq += psq[o];
      m   = fmaxf(m, pmax[o]);
    }
    const float N = (float)HW;
    const float mean = s / N;
    const float var  = (sq - s * s / N) / (N - 1.0f);   // unbiased (ddof=1)
    feats[t]           = mean;
    feats[NCH + t]     = m;
    feats[2 * NCH + t] = sqrtf(fmaxf(var, 0.f));
  }
  __syncthreads();

  // ---- Phase 2: layer 1 via v_wmma_f32_16x16x32_f16 (waves 0..3) ----
  const int wv   = t >> 5;
  const int lane = t & 31;
  if (wv < 4) {
    const int n0 = wv * 16;          // this wave's W1 column tile
    const int g8 = (lane >> 4) << 3; // K offset for lane group (0 or 8)
    const int n  = lane & 15;        // column within tile

    v8f acc = {};
    for (int kt = 0; kt < 36; ++kt) {
      const int kb = kt * 32 + g8;
      v16h a, b;
#pragma unroll
      for (int j = 0; j < 8; ++j) {
        // 16-bit A/B K striping: VGPR j<4 -> K = 2j(+1), j>=4 -> K = 16+2(j-4)(+1)
        const int k0 = kb + ((j < 4) ? (2 * j) : (8 + 2 * j));
        a[2 * j]     = (_Float16)feats[k0];
        a[2 * j + 1] = (_Float16)feats[k0 + 1];
        b[2 * j]     = (_Float16)W1[(long)k0 * 64 + n0 + n];
        b[2 * j + 1] = (_Float16)W1[(long)(k0 + 1) * 64 + n0 + n];
      }
      acc = __builtin_amdgcn_wmma_f32_16x16x32_f16(
          /*neg_a=*/false, a, /*neg_b=*/false, b,
          /*c_mod=*/(short)0, acc, /*reuse_a=*/false, /*reuse_b=*/false);
    }
    // D row M=0 lives in acc[0] of lanes 0..15 (N = lane)
    if (lane < 16) {
      h1[n0 + lane] = fmaxf(acc[0] + b1[n0 + lane], 0.f);
    }
  }
  __syncthreads();

  // ---- Phase 3: layer 2 (64 -> 16), relu ----
  if (t < 16) {
    float a2 = b2[t];
    for (int k = 0; k < 64; ++k) a2 += h1[k] * W2[k * 16 + t];
    h2v[t] = fmaxf(a2, 0.f);
  }
  __syncthreads();

  // ---- layer 3 (16 -> 2) + softmax ----
  if (t == 0) {
    float l0 = b3[0], l1 = b3[1];
    for (int k = 0; k < 16; ++k) {
      l0 += h2v[k] * W3[k * 2 + 0];
      l1 += h2v[k] * W3[k * 2 + 1];
    }
    const float m  = fmaxf(l0, l1);
    const float e0 = __expf(l0 - m);
    const float e1 = __expf(l1 - m);
    const float inv = 1.0f / (e0 + e1);
    out[0] = e0 * inv;
    out[1] = e1 * inv;
  }
}

// ---------------------------------------------------------------------------
extern "C" void kernel_launch(void* const* d_in, const int* in_sizes, int n_in,
                              void* d_out, int out_size, void* d_ws, size_t ws_size,
                              hipStream_t stream) {
  (void)in_sizes; (void)n_in; (void)out_size;

  const f4*    x4 = (const f4*)d_in[0];
  const float* W1 = (const float*)d_in[1];
  const float* b1 = (const float*)d_in[2];
  const float* W2 = (const float*)d_in[3];
  const float* b2 = (const float*)d_in[4];
  const float* W3 = (const float*)d_in[5];
  const float* b3 = (const float*)d_in[6];
  float* out = (float*)d_out;
  float* ws  = (float*)d_ws;

  // Deterministic workspace-aware block count: NB * 384 * 3 floats of partials.
  int NB = 2048;
  while (NB > 1 && (size_t)NB * NCH * 3 * sizeof(float) > ws_size) NB >>= 1;
  const int rows_per_block = HW / NB;   // NB is a power of two dividing 2^20

  float* psum = ws;
  float* psq  = ws + (size_t)NB * NCH;
  float* pmax = ws + (size_t)2 * NB * NCH;

  pool_partial_kernel<<<dim3(NB), dim3(NCH), 0, stream>>>(
      x4, psum, psq, pmax, rows_per_block);

  finalize_mlp_kernel<<<dim3(1), dim3(NCH), 0, stream>>>(
      psum, psq, pmax, NB, W1, b1, W2, b2, W3, b3, out);
}
